// ModelQAT_85959475462475
// MI455X (gfx1250) — matrix operand order repaired and checked by reading, not compiled, not run on hardware
//
#include <hip/hip_runtime.h>
#include <math.h>

// ---------------- problem constants (match reference) ----------------
#define N_NODES 100000   // multiple of 16
#define N_EDGES 1200000  // multiple of 16
#define CH      64
#define NLAYERS 3
#define NGRAPH  512
#define X_DIM   16
#define EA_DIM  8
#define NEG_SLOPE 0.01f

typedef __attribute__((ext_vector_type(16))) _Float16 v16h;
typedef __attribute__((ext_vector_type(8)))  float    v8f;

__device__ __forceinline__ float lrelu_f(float v) { return v > 0.f ? v : NEG_SLOPE * v; }

// ---------------- elementwise / fill ----------------
__global__ void fill_f32_k(float* p, float v, long n) {
    for (long i = blockIdx.x * (long)blockDim.x + threadIdx.x; i < n;
         i += (long)gridDim.x * blockDim.x) p[i] = v;
}

// ---------------- per-tensor absmax (fake-quant scale) ----------------
__global__ void absmax_k(const float* __restrict__ x, long n, float* slot) {
    __shared__ float sm[256];
    float m = 0.f;
    for (long i = blockIdx.x * (long)blockDim.x + threadIdx.x; i < n;
         i += (long)gridDim.x * blockDim.x) m = fmaxf(m, fabsf(x[i]));
    sm[threadIdx.x] = m;
    __syncthreads();
    for (int s = blockDim.x >> 1; s > 0; s >>= 1) {
        if (threadIdx.x < (unsigned)s) sm[threadIdx.x] = fmaxf(sm[threadIdx.x], sm[threadIdx.x + s]);
        __syncthreads();
    }
    if (threadIdx.x == 0)
        atomicMax((unsigned int*)slot, __float_as_uint(sm[0]));  // vals >= 0: uint order == float order
}

// fake-quant weights -> f16, TRANSPOSED and K-padded:  out[c * KP + k]
__global__ void quant_wt_k(const float* __restrict__ w, int K, int Nc, int KP,
                           const float* slot, float q, _Float16* __restrict__ out) {
    float s = slot[0] / q + 1e-12f;
    float inv = 1.f / s;
    long n = (long)K * Nc;
    for (long i = blockIdx.x * (long)blockDim.x + threadIdx.x; i < n;
         i += (long)gridDim.x * blockDim.x) {
        int k = (int)(i / Nc), c = (int)(i % Nc);
        float r = rintf(w[i] * inv);
        r = fminf(fmaxf(r, -q), q);
        out[(long)c * KP + k] = (_Float16)(r * s);
    }
}

// fake-quant activations -> f32, row-padded to DP columns: out[row * DP + d]
__global__ void quant_a_k(const float* __restrict__ x, long rows, int D, int DP,
                          const float* slot, float q, float* __restrict__ out) {
    float s = slot[0] / q + 1e-12f;
    float inv = 1.f / s;
    long n = rows * D;
    for (long i = blockIdx.x * (long)blockDim.x + threadIdx.x; i < n;
         i += (long)gridDim.x * blockDim.x) {
        long r0 = i / D; int d = (int)(i % D);
        float r = rintf(x[i] * inv);
        r = fminf(fmaxf(r, -q), q);
        out[r0 * DP + d] = r * s;
    }
}

// ---------------- WMMA GEMM:  Y[M,Nc] = act( X[M,KP] * Wt + bias ) (+res) ----------------
// X: f32, row stride KP (zero-padded if K<KP), converted to f16 in-register.
// Wt: f16, transposed [Nc][KP], zero-padded.  Y: f32, stride Nc = NT*16.  M % 16 == 0.
// One wave -> one 16-row strip across ALL NT column tiles (X read exactly once).
// 8 waves/block -> 128 rows/block; grid.x only.  RES compile-time => branch-free epilogue.
template <int KP, int NT, bool RES>
__global__ __launch_bounds__(256) void gemm_wmma_k(
        const float* __restrict__ X, const _Float16* __restrict__ Wt,
        const float* __restrict__ bias, float* __restrict__ Y,
        const float* __restrict__ res, int M, int act) {
    constexpr int Nc = NT * 16;
    const int lane = threadIdx.x & 31;
    const int wave = threadIdx.x >> 5;
    const int row0 = (blockIdx.x * 8 + wave) * 16;   // wave-uniform
    if (row0 >= M) return;                           // uniform -> EXEC all ones at WMMA

    const int half = lane >> 4;
    const int l15  = lane & 15;
    // A: lane covers row (row0+l15); 16 contiguous K values at kb + half*16.
    const float* Xr = X + (long)(row0 + l15) * KP + half * 16;
    // B: lane covers col (t*16 + l15); 16 contiguous K values at kb + half*16.
    const _Float16* Wc0 = Wt + (long)l15 * KP + half * 16;

    v8f acc[NT];
#pragma unroll
    for (int t = 0; t < NT; ++t) acc[t] = (v8f){};

#pragma unroll
    for (int kb = 0; kb < KP; kb += 32) {
        v8f x0 = *(const v8f*)(Xr + kb);          // 32B-aligned
        v8f x1 = *(const v8f*)(Xr + kb + 8);
        v16h a;
#pragma unroll
        for (int j = 0; j < 8; ++j) {
            a[j]     = (_Float16)x0[j];
            a[8 + j] = (_Float16)x1[j];
        }
#pragma unroll
        for (int t = 0; t < NT; ++t) {
            v16h b = *(const v16h*)(Wc0 + (long)t * 16 * KP + kb);   // L0-resident
            acc[t] = __builtin_amdgcn_wmma_f32_16x16x32_f16(false, a, false, b,
                                                            (short)0, acc[t], false, false);
        }
    }
    // epilogue: lane holds rows (row0 + v + 8*half), col (t*16 + l15)
    const long mbase = (long)(row0 + half * 8) * Nc + l15;
#pragma unroll
    for (int t = 0; t < NT; ++t) {
        const int colw = t * 16;
        const float bv = bias[colw + l15];
        float rv[8];
        if (RES) {
#pragma unroll
            for (int v = 0; v < 8; ++v) rv[v] = res[mbase + (long)v * Nc + colw];
        }
#pragma unroll
        for (int v = 0; v < 8; ++v) {
            float val = acc[t][v] + bv;
            if (act == 1) val = lrelu_f(val);
            if (RES) val += rv[v];
            Y[mbase + (long)v * Nc + colw] = val;
        }
    }
}

static void launch_gemm(int KP, const float* X, const _Float16* Wt, const float* bias,
                        float* Y, const float* res, int M, int Nc, int act, hipStream_t s) {
    dim3 grid((M + 127) / 128);
    if (KP == 32 && Nc == 64)
        gemm_wmma_k<32, 4, false><<<grid, 256, 0, s>>>(X, Wt, bias, Y, res, M, act);
    else if (KP == 64 && Nc == 64)
        gemm_wmma_k<64, 4, false><<<grid, 256, 0, s>>>(X, Wt, bias, Y, res, M, act);
    else if (KP == 64 && Nc == 128)
        gemm_wmma_k<64, 8, false><<<grid, 256, 0, s>>>(X, Wt, bias, Y, res, M, act);
    else if (res)
        gemm_wmma_k<128, 4, true><<<grid, 256, 0, s>>>(X, Wt, bias, Y, res, M, act);
    else
        gemm_wmma_k<128, 4, false><<<grid, 256, 0, s>>>(X, Wt, bias, Y, res, M, act);
}

// ---------------- BatchNorm (training-mode batch stats) ----------------
__global__ void bn_stats_k(const float* __restrict__ X, long M, int C,
                           float* __restrict__ sum, float* __restrict__ sumsq) {
    __shared__ float s1[128], s2[128];
    for (int i = threadIdx.x; i < C; i += blockDim.x) { s1[i] = 0.f; s2[i] = 0.f; }
    __syncthreads();
    long total = M * C;
    for (long i = blockIdx.x * (long)blockDim.x + threadIdx.x; i < total;
         i += (long)gridDim.x * blockDim.x) {
        int c = (int)(i & (C - 1));   // C is 64 or 128
        float v = X[i];
        atomicAdd(&s1[c], v);
        atomicAdd(&s2[c], v * v);
    }
    __syncthreads();
    for (int i = threadIdx.x; i < C; i += blockDim.x) {
        atomicAdd(&sum[i], s1[i]);
        atomicAdd(&sumsq[i], s2[i]);
    }
}

__global__ void bn_final_k(const float* sum, const float* sumsq, const float* g,
                           const float* b, float invM, int C, float* scale, float* shift) {
    int c = blockIdx.x * blockDim.x + threadIdx.x;
    if (c < C) {
        float mu  = sum[c] * invM;
        float var = sumsq[c] * invM - mu * mu;
        float sc  = g[c] * rsqrtf(var + 1e-5f);
        scale[c] = sc;
        shift[c] = b[c] - mu * sc;
    }
}

__global__ void bn_apply_k(float* __restrict__ X, long M, int C,
                           const float* scale, const float* shift, int act) {
    long total = M * C;
    for (long i = blockIdx.x * (long)blockDim.x + threadIdx.x; i < total;
         i += (long)gridDim.x * blockDim.x) {
        int c = (int)(i & (C - 1));
        float v = X[i] * scale[c] + shift[c];
        if (act == 1) v = lrelu_f(v);
        X[i] = v;
    }
}

// ---------------- GENConv softmax aggregation ----------------
__device__ __forceinline__ void atomicMaxF(float* addr, float val) {
    if (val >= 0.f) atomicMax((int*)addr, __float_as_int(val));
    else            atomicMin((unsigned int*)addr, __float_as_uint(val));
}

__global__ void gen_max_k(const float* __restrict__ h, const float* __restrict__ ea,
                          const int* __restrict__ src, const int* __restrict__ dst,
                          const float* tptr, int tidx, float* __restrict__ m) {
    float t = tptr[tidx];
    long total = (long)N_EDGES * CH;
    for (long i = blockIdx.x * (long)blockDim.x + threadIdx.x; i < total;
         i += (long)gridDim.x * blockDim.x) {
        long e = i >> 6; int c = (int)(i & 63);
        float msg = fmaxf(h[(long)src[e] * CH + c] + ea[i], 0.f) + 1e-7f;
        atomicMaxF(&m[(long)dst[e] * CH + c], msg * t);
    }
}

__global__ void gen_finm_k(float* m, long n) {
    for (long i = blockIdx.x * (long)blockDim.x + threadIdx.x; i < n;
         i += (long)gridDim.x * blockDim.x) {
        float v = m[i];
        if (!isfinite(v)) m[i] = 0.f;
    }
}

__global__ void gen_sum_k(const float* __restrict__ h, const float* __restrict__ ea,
                          const int* __restrict__ src, const int* __restrict__ dst,
                          const float* tptr, int tidx, const float* __restrict__ m,
                          float* __restrict__ denom, float* __restrict__ num) {
    float t = tptr[tidx];
    long total = (long)N_EDGES * CH;
    for (long i = blockIdx.x * (long)blockDim.x + threadIdx.x; i < total;
         i += (long)gridDim.x * blockDim.x) {
        long e = i >> 6; int c = (int)(i & 63);
        long di = (long)dst[e] * CH + c;
        float msg = fmaxf(h[(long)src[e] * CH + c] + ea[i], 0.f) + 1e-7f;
        float ex  = __expf(msg * t - m[di]);
        atomicAdd(&denom[di], ex);
        atomicAdd(&num[di], ex * msg);
    }
}

__global__ void gen_upd_k(const float* __restrict__ h, const float* __restrict__ num,
                          const float* __restrict__ denom, float* __restrict__ out, long n) {
    for (long i = blockIdx.x * (long)blockDim.x + threadIdx.x; i < n;
         i += (long)gridDim.x * blockDim.x)
        out[i] = h[i] + num[i] / (denom[i] + 1e-16f);
}

// ---------------- global mean pool ----------------
__global__ void pool_acc_k(const float* __restrict__ h, const int* __restrict__ batch,
                           float* __restrict__ pooled, float* __restrict__ counts) {
    long total = (long)N_NODES * CH;
    for (long i = blockIdx.x * (long)blockDim.x + threadIdx.x; i < total;
         i += (long)gridDim.x * blockDim.x) {
        long n = i >> 6; int c = (int)(i & 63);
        int g = batch[n];
        atomicAdd(&pooled[(long)g * CH + c], h[i]);
        if (c == 0) atomicAdd(&counts[g], 1.f);
    }
}

__global__ void pool_fin_k(float* pooled, const float* counts) {
    long total = (long)NGRAPH * CH;
    for (long i = blockIdx.x * (long)blockDim.x + threadIdx.x; i < total;
         i += (long)gridDim.x * blockDim.x) {
        long g = i >> 6;
        pooled[i] /= fmaxf(counts[g], 1.f);
    }
}

// ---------------- final 64 -> 1 linear + sigmoid ----------------
__global__ void head_fin_k(const float* __restrict__ o, const _Float16* __restrict__ wf,
                           const float* bf, float* __restrict__ out) {
    int g = blockIdx.x * blockDim.x + threadIdx.x;
    if (g < NGRAPH) {
        float s = bf[0];
#pragma unroll
        for (int c = 0; c < CH; ++c) s += o[(long)g * CH + c] * (float)wf[c];
        out[g] = 1.f / (1.f + __expf(-s));
    }
}

// ======================= host orchestration =======================
static inline int gsz(long n, int t, int cap) {
    long b = (n + t - 1) / t;
    return (int)(b > cap ? cap : (b < 1 ? 1 : b));
}

extern "C" void kernel_launch(void* const* d_in, const int* in_sizes, int n_in,
                              void* d_out, int out_size, void* d_ws, size_t ws_size,
                              hipStream_t stream) {
    (void)in_sizes; (void)n_in; (void)out_size; (void)ws_size;
    const float* x    = (const float*)d_in[0];
    const float* eatt = (const float*)d_in[1];
    const int*   eidx = (const int*)d_in[2];
    const int*   batch= (const int*)d_in[3];
    const float* Wn0  = (const float*)d_in[4];  const float* bn0 = (const float*)d_in[5];
    const float* WnH  = (const float*)d_in[6];  const float* bnH = (const float*)d_in[7];
    const float* We0  = (const float*)d_in[8];  const float* be0 = (const float*)d_in[9];
    const float* WeH  = (const float*)d_in[10]; const float* beH = (const float*)d_in[11];
    const float* g_node=(const float*)d_in[12]; const float* b_node=(const float*)d_in[13];
    const float* g_edge=(const float*)d_in[14]; const float* b_edge=(const float*)d_in[15];
    const float* conv_t=(const float*)d_in[16];
    const float* mW1  = (const float*)d_in[17]; const float* mb1 = (const float*)d_in[18];
    const float* mg   = (const float*)d_in[19]; const float* mbe = (const float*)d_in[20];
    const float* mW2  = (const float*)d_in[21]; const float* mb2 = (const float*)d_in[22];
    const float* eW1  = (const float*)d_in[23]; const float* eb1 = (const float*)d_in[24];
    const float* eg   = (const float*)d_in[25]; const float* ebe = (const float*)d_in[26];
    const float* eW2  = (const float*)d_in[27]; const float* eb2 = (const float*)d_in[28];
    const float* oWh  = (const float*)d_in[29]; const float* obh = (const float*)d_in[30];
    const float* oWf  = (const float*)d_in[31]; const float* obf = (const float*)d_in[32];
    float* out = (float*)d_out;

    const int* src = eidx;
    const int* dst = eidx + N_EDGES;

    // ---- workspace carve-up ----
    char* base = (char*)d_ws;
    size_t off = 0;
    auto alloc = [&](size_t bytes) -> char* {
        char* p = base + off;
        off += (bytes + 255) & ~(size_t)255;
        return p;
    };
    float* scales = (float*)alloc(32 * 4);
    float* bnsum  = (float*)alloc(256 * 4);
    float* bnsc   = (float*)alloc(256 * 4);
    // transposed, K-padded f16 weights: [Nc][KP]
    _Float16* qWn0 = (_Float16*)alloc((size_t)CH * 32 * 2);            // K=16 -> KP=32
    _Float16* qWnH = (_Float16*)alloc((size_t)2 * CH * CH * 2);        // KP=64
    _Float16* qWe0 = (_Float16*)alloc((size_t)CH * 32 * 2);            // K=8  -> KP=32
    _Float16* qWeH = (_Float16*)alloc((size_t)2 * CH * CH * 2);
    _Float16* qmW1 = (_Float16*)alloc((size_t)NLAYERS * 2 * CH * CH * 2);  // [128][64]
    _Float16* qmW2 = (_Float16*)alloc((size_t)NLAYERS * CH * 2 * CH * 2);  // [64][128]
    _Float16* qeW1 = (_Float16*)alloc((size_t)NLAYERS * 2 * CH * CH * 2);
    _Float16* qeW2 = (_Float16*)alloc((size_t)NLAYERS * CH * 2 * CH * 2);
    _Float16* qoWh = (_Float16*)alloc((size_t)2 * CH * CH * 2);
    _Float16* qoWf = (_Float16*)alloc((size_t)CH * 2);                 // [1][64]
    // activations (inputs row-padded to 32)
    float* xq     = (float*)alloc((size_t)N_NODES * 32 * 4);
    float* eaq    = (float*)alloc((size_t)N_EDGES * 32 * 4);
    float* hA     = (float*)alloc((size_t)N_NODES * CH * 4);
    float* hB     = (float*)alloc((size_t)N_NODES * CH * 4);
    float* h128   = (float*)alloc((size_t)N_NODES * 2 * CH * 4);
    float* eA     = (float*)alloc((size_t)N_EDGES * CH * 4);
    float* eB     = (float*)alloc((size_t)N_EDGES * CH * 4);
    float* e128   = (float*)alloc((size_t)N_EDGES * 2 * CH * 4);
    float* mbuf   = (float*)alloc((size_t)N_NODES * CH * 4);
    float* dbuf   = (float*)alloc((size_t)N_NODES * CH * 4);
    float* nbuf   = (float*)alloc((size_t)N_NODES * CH * 4);
    float* pooled = (float*)alloc((size_t)NGRAPH * CH * 4);
    float* counts = (float*)alloc((size_t)NGRAPH * 4);
    float* o1     = (float*)alloc((size_t)NGRAPH * CH * 4);
    float* o2     = (float*)alloc((size_t)NGRAPH * CH * 4);

    const int T = 256;
    auto fill = [&](float* p, float v, long n) {
        fill_f32_k<<<gsz(n, T, 4096), T, 0, stream>>>(p, v, n);
    };
    auto amax = [&](const float* p, long n, int slot) {
        absmax_k<<<gsz(n, T, 1024), T, 0, stream>>>(p, n, scales + slot);
    };
    auto qwt = [&](const float* p, int K, int Nc, int KP, int slot, float q, _Float16* o) {
        quant_wt_k<<<gsz((long)K * Nc, T, 1024), T, 0, stream>>>(p, K, Nc, KP, scales + slot, q, o);
    };
    auto bn = [&](float* X, long M, int C, const float* g, const float* b, int act) {
        fill(bnsum, 0.f, 256);
        bn_stats_k<<<gsz(M * C, T, 2048), T, 0, stream>>>(X, M, C, bnsum, bnsum + 128);
        bn_final_k<<<1, 128, 0, stream>>>(bnsum, bnsum + 128, g, b, 1.f / (float)M, C,
                                          bnsc, bnsc + 128);
        bn_apply_k<<<gsz(M * C, T, 4096), T, 0, stream>>>(X, M, C, bnsc, bnsc + 128, act);
    };

    // ---- phase 0: zero padded buffers ----
    fill(scales, 0.f, 32);
    fill(xq, 0.f, (long)N_NODES * 32);
    fill(eaq, 0.f, (long)N_EDGES * 32);
    fill((float*)qWn0, 0.f, (long)CH * 32 / 2);
    fill((float*)qWe0, 0.f, (long)CH * 32 / 2);

    // ---- phase 1: absmax scales ----
    amax(x, (long)N_NODES * X_DIM, 0);
    amax(eatt, (long)N_EDGES * EA_DIM, 1);
    amax(Wn0, X_DIM * CH, 2);
    amax(WnH, CH * CH, 3); amax(WnH + CH * CH, CH * CH, 4);
    amax(We0, EA_DIM * CH, 5);
    amax(WeH, CH * CH, 6); amax(WeH + CH * CH, CH * CH, 7);
    for (int i = 0; i < NLAYERS; ++i) {
        amax(mW1 + (long)i * CH * 2 * CH, CH * 2 * CH, 8 + i);
        amax(mW2 + (long)i * 2 * CH * CH, 2 * CH * CH, 11 + i);
        amax(eW1 + (long)i * CH * 2 * CH, CH * 2 * CH, 14 + i);
        amax(eW2 + (long)i * 2 * CH * CH, 2 * CH * CH, 17 + i);
    }
    amax(oWh, CH * CH, 20); amax(oWh + CH * CH, CH * CH, 21);
    amax(oWf, CH, 22);

    // ---- phase 2: quantize (first encoder layer 8-bit, rest 4-bit) ----
    quant_a_k<<<gsz((long)N_NODES * X_DIM, T, 2048), T, 0, stream>>>(
        x, N_NODES, X_DIM, 32, scales + 0, 7.f, xq);
    quant_a_k<<<gsz((long)N_EDGES * EA_DIM, T, 4096), T, 0, stream>>>(
        eatt, N_EDGES, EA_DIM, 32, scales + 1, 7.f, eaq);
    qwt(Wn0, X_DIM, CH, 32, 2, 127.f, qWn0);
    qwt(WnH, CH, CH, CH, 3, 7.f, qWnH);
    qwt(WnH + CH * CH, CH, CH, CH, 4, 7.f, qWnH + CH * CH);
    qwt(We0, EA_DIM, CH, 32, 5, 127.f, qWe0);
    qwt(WeH, CH, CH, CH, 6, 7.f, qWeH);
    qwt(WeH + CH * CH, CH, CH, CH, 7, 7.f, qWeH + CH * CH);
    for (int i = 0; i < NLAYERS; ++i) {
        qwt(mW1 + (long)i * CH * 2 * CH, CH, 2 * CH, CH, 8 + i,  7.f, qmW1 + (long)i * 2 * CH * CH);
        qwt(mW2 + (long)i * 2 * CH * CH, 2 * CH, CH, 2 * CH, 11 + i, 7.f, qmW2 + (long)i * CH * 2 * CH);
        qwt(eW1 + (long)i * CH * 2 * CH, CH, 2 * CH, CH, 14 + i, 7.f, qeW1 + (long)i * 2 * CH * CH);
        qwt(eW2 + (long)i * 2 * CH * CH, 2 * CH, CH, 2 * CH, 17 + i, 7.f, qeW2 + (long)i * CH * 2 * CH);
    }
    qwt(oWh, CH, CH, CH, 20, 7.f, qoWh);
    qwt(oWh + CH * CH, CH, CH, CH, 21, 7.f, qoWh + CH * CH);
    qwt(oWf, CH, 1, CH, 22, 7.f, qoWf);

    // ---- phase 3: encoders (WMMA) ----
    launch_gemm(32, xq, qWn0, bn0, hA, nullptr, N_NODES, CH, 1, stream);
    launch_gemm(64, hA, qWnH,           bnH,      hB, nullptr, N_NODES, CH, 1, stream);
    launch_gemm(64, hB, qWnH + CH * CH, bnH + CH, hA, nullptr, N_NODES, CH, 1, stream);
    launch_gemm(32, eaq, qWe0, be0, eA, nullptr, N_EDGES, CH, 1, stream);
    launch_gemm(64, eA, qWeH,           beH,      eB, nullptr, N_EDGES, CH, 1, stream);
    launch_gemm(64, eB, qWeH + CH * CH, beH + CH, eA, nullptr, N_EDGES, CH, 1, stream);

    // ---- phase 4: bn_input ----
    bn(hA, N_NODES, CH, g_node, b_node, 0);
    bn(eA, N_EDGES, CH, g_edge, b_edge, 0);

    // ---- phase 5: message-passing layers ----
    float* hcur = hA;
    float* ecur = eA;
    float* eoth = eB;
    const long NC = (long)N_NODES * CH;
    for (int i = 0; i < NLAYERS; ++i) {
        fill(mbuf, -INFINITY, NC);
        fill(dbuf, 0.f, NC);
        fill(nbuf, 0.f, NC);
        gen_max_k<<<8192, T, 0, stream>>>(hcur, ecur, src, dst, conv_t, i, mbuf);
        gen_finm_k<<<gsz(NC, T, 4096), T, 0, stream>>>(mbuf, NC);
        gen_sum_k<<<8192, T, 0, stream>>>(hcur, ecur, src, dst, conv_t, i, mbuf, dbuf, nbuf);
        gen_upd_k<<<gsz(NC, T, 4096), T, 0, stream>>>(hcur, nbuf, dbuf, hB, NC);
        // node MLP: 64 -> 128 (bn+lrelu) -> 64, residual fused into 2nd GEMM epilogue
        launch_gemm(64, hB, qmW1 + (long)i * 2 * CH * CH, mb1 + (long)i * 2 * CH,
                    h128, nullptr, N_NODES, 2 * CH, 0, stream);
        bn(h128, N_NODES, 2 * CH, mg + (long)i * 2 * CH, mbe + (long)i * 2 * CH, 1);
        launch_gemm(128, h128, qmW2 + (long)i * CH * 2 * CH, mb2 + (long)i * CH,
                    hcur, hcur /*residual = skip*/, N_NODES, CH, 0, stream);
        // edge MLP: ea = mlp(ea)
        launch_gemm(64, ecur, qeW1 + (long)i * 2 * CH * CH, eb1 + (long)i * 2 * CH,
                    e128, nullptr, N_EDGES, 2 * CH, 0, stream);
        bn(e128, N_EDGES, 2 * CH, eg + (long)i * 2 * CH, ebe + (long)i * 2 * CH, 1);
        launch_gemm(128, e128, qeW2 + (long)i * CH * 2 * CH, eb2 + (long)i * CH,
                    eoth, nullptr, N_EDGES, CH, 0, stream);
        float* t = ecur; ecur = eoth; eoth = t;
    }

    // ---- phase 6: global mean pool ----
    fill(pooled, 0.f, (long)NGRAPH * CH);
    fill(counts, 0.f, NGRAPH);
    pool_acc_k<<<gsz(NC, T, 4096), T, 0, stream>>>(hcur, batch, pooled, counts);
    pool_fin_k<<<gsz((long)NGRAPH * CH, T, 512), T, 0, stream>>>(pooled, counts);

    // ---- phase 7: output head (WMMA) + sigmoid ----
    launch_gemm(64, pooled, qoWh,           obh,      o1, nullptr, NGRAPH, CH, 1, stream);
    launch_gemm(64, o1,     qoWh + CH * CH, obh + CH, o2, nullptr, NGRAPH, CH, 1, stream);
    head_fin_k<<<(NGRAPH + T - 1) / T, T, 0, stream>>>(o2, qoWf, obf, out);
}
